// GraphormerLayer_17489106829950
// MI455X (gfx1250) — matrix operand order repaired and checked by reading, not compiled, not run on hardware
//
#include <hip/hip_runtime.h>
#include <hip/hip_bf16.h>
#include <stdint.h>
#include <math.h>

// ---------------------------------------------------------------------------
// Problem constants (match reference)
// ---------------------------------------------------------------------------
#define H_DIM   768
#define NHEAD   12
#define HDIM    64
#define FF_DIM  3072
#define BATCH   8
#define SEQ     1024
#define MROWS   (BATCH*SEQ)

// ---------------------------------------------------------------------------
// WMMA plumbing (gfx1250, wave32):  D = A(16x32 bf16) x B(32x16 bf16) + C(f32)
// ---------------------------------------------------------------------------
typedef __attribute__((ext_vector_type(16))) __bf16 v16bf;
typedef __attribute__((ext_vector_type(8)))  float  v8f;

union Frag { v16bf v; uint32_t u[8]; };

__device__ __forceinline__ v8f wmma_bf16(const Frag& a, const Frag& b, v8f c) {
    return __builtin_amdgcn_wmma_f32_16x16x32_bf16(
        /*neg_a=*/false, a.v, /*neg_b=*/false, b.v,
        /*c_mod=*/(short)0, c, /*reuse_a=*/false, /*reuse_b=*/false);
}

// LDS byte offset of a __shared__ object: group-segment offset lives in the
// low 32 bits of the generic address (LDS aperture layout).
__device__ __forceinline__ uint32_t lds_off(const void* p) {
    return (uint32_t)(uintptr_t)p;
}

// ---------------------------------------------------------------------------
// fp32 -> bf16 conversion (activations)
// ---------------------------------------------------------------------------
__global__ void cvt_f32_bf16_kernel(const float* __restrict__ in,
                                    __bf16* __restrict__ out, int n) {
    int i = blockIdx.x * blockDim.x + threadIdx.x;
    if (i < n) out[i] = (__bf16)in[i];
}

// ---------------------------------------------------------------------------
// fp32 [K][N] -> bf16 transposed [N][K] (weights), tiled 32x32 via LDS.
// Lets the GEMM use identical row-major async block-copies for A and W^T.
// ---------------------------------------------------------------------------
__launch_bounds__(256)
__global__ void cvt_transpose_kernel(const float* __restrict__ in,
                                     __bf16* __restrict__ out, int K, int N) {
    __shared__ __bf16 tile[32][33];
    const int kb = blockIdx.y * 32;
    const int nb = blockIdx.x * 32;
    const int tx = threadIdx.x & 31;
    const int ty = threadIdx.x >> 5;        // 0..7
    for (int i = ty; i < 32; i += 8)
        tile[i][tx] = (__bf16)in[(size_t)(kb + i) * N + nb + tx];
    __syncthreads();
    for (int i = ty; i < 32; i += 8)
        out[(size_t)(nb + i) * K + kb + tx] = tile[tx][i];
}

// ---------------------------------------------------------------------------
// Async 128x32 bf16 tile copy: global (row-major, leading dim ld) -> LDS
// (padded stride A_LD). Each thread issues 2x GLOBAL_LOAD_ASYNC_TO_LDS_B128
// (16B/lane). Tracked by ASYNCcnt; caller waits + barriers.
// ---------------------------------------------------------------------------
#define TB_M 128
#define TB_N 128
#define TB_K 32
#define A_LD 40   // 80B row stride: 16B-aligned for B128, 20-bank step (conflict-free)

__device__ __forceinline__ void async_tile_128x32(const __bf16* __restrict__ g,
                                                  size_t row0, int ld, int k0,
                                                  __bf16* lds, int tid)
{
    for (int it = 0; it < 2; ++it) {
        int c   = tid + it * 256;          // 512 chunks of 16B
        int row = c >> 2;                  // 4 chunks per 64B row
        int col = (c & 3) * 8;
        const __bf16* gp = g + (row0 + row) * (size_t)ld + k0 + col;
        uint32_t lo = lds_off(&lds[row * A_LD + col]);
        asm volatile("global_load_async_to_lds_b128 %0, %1, off"
                     :: "v"(lo), "v"(gp) : "memory");
    }
}

__device__ __forceinline__ void wait_async_all() {
    asm volatile("s_wait_asynccnt 0" ::: "memory");
}

// ---------------------------------------------------------------------------
// Tiled WMMA GEMM:  out = epilogue(A[M,K] @ W[K,N] + bias[N])
//   A bf16 [M][K]; Wt bf16 [N][K] (pre-transposed). Accumulate f32.
//   Block tile 128x128x32; 8 waves (4 in M, 2 in N); each wave 32x64.
//   Double-buffered async global->LDS copies overlap WMMA with the fill.
// ---------------------------------------------------------------------------
__launch_bounds__(256)
__global__ void gemm_bf16_kernel(const __bf16* __restrict__ A,
                                 const __bf16* __restrict__ Wt,   // [N][K]
                                 const float* __restrict__ bias,
                                 const float* __restrict__ residual, // nullable
                                 float* __restrict__ outF,           // nullable
                                 __bf16* __restrict__ outB,          // nullable
                                 int M, int K, int N, int doGelu)
{
    __shared__ __align__(16) __bf16 Abuf[2][TB_M * A_LD];
    __shared__ __align__(16) __bf16 Wbuf[2][TB_N * A_LD];

    const int tid   = threadIdx.x;
    const int lane  = tid & 31;
    const int wave  = tid >> 5;
    const int waveM = wave & 3;            // 0..3
    const int waveN = wave >> 2;           // 0..1
    const int blockM = blockIdx.y * TB_M;
    const int blockN = blockIdx.x * TB_N;
    const int halfw = lane >> 4;           // 0/1
    const int l16   = lane & 15;

    v8f acc[2][4];
    for (int i = 0; i < 2; ++i)
        for (int j = 0; j < 4; ++j)
            for (int e = 0; e < 8; ++e) acc[i][j][e] = 0.0f;

    // prologue: fill buffer 0
    async_tile_128x32(A,  (size_t)blockM, K, 0, Abuf[0], tid);
    async_tile_128x32(Wt, (size_t)blockN, K, 0, Wbuf[0], tid);
    wait_async_all();
    __syncthreads();

    const int nK = K / TB_K;
    for (int kt = 0; kt < nK; ++kt) {
        const int cur = kt & 1;
        const int nxt = cur ^ 1;
        if (kt + 1 < nK) {
            int k0 = (kt + 1) * TB_K;
            async_tile_128x32(A,  (size_t)blockM, K, k0, Abuf[nxt], tid);
            async_tile_128x32(Wt, (size_t)blockN, K, k0, Wbuf[nxt], tid);
        }

        const __bf16* Al = Abuf[cur];
        const __bf16* Wl = Wbuf[cur];

        Frag af[2];
        for (int tm = 0; tm < 2; ++tm) {
            int rowB = waveM * 32 + tm * 16 + l16;      // A row = lane&15
            for (int r = 0; r < 8; ++r) {
                // 16-bit A layout: v0-3 K0-7 / K8-15, v4-7 K16-23 / K24-31
                int kf = 16 * (r >> 2) + 8 * halfw + 2 * (r & 3);
                af[tm].u[r] = *(const uint32_t*)&Al[rowB * A_LD + kf];
            }
        }
        for (int tn = 0; tn < 4; ++tn) {
            Frag bf;
            int nB = waveN * 64 + tn * 16 + l16;        // B col = lane&15
            for (int r = 0; r < 8; ++r) {
                int kf = 16 * (r >> 2) + 8 * halfw + 2 * (r & 3);   // v0-3 K0-7 / K8-15, v4-7 K16-23 / K24-31
                bf.u[r] = *(const uint32_t*)&Wl[nB * A_LD + kf];
            }
            acc[0][tn] = wmma_bf16(af[0], bf, acc[0][tn]);
            acc[1][tn] = wmma_bf16(af[1], bf, acc[1][tn]);
        }

        wait_async_all();       // next buffer landed
        __syncthreads();        // everyone done reading cur
    }

    // ---- epilogue: bias (+gelu) (+residual); f32 and/or bf16 stores ----
    for (int tm = 0; tm < 2; ++tm)
        for (int tn = 0; tn < 4; ++tn)
            for (int r = 0; r < 8; ++r) {
                int row = blockM + waveM * 32 + tm * 16 + 8 * halfw + r;
                int col = blockN + waveN * 64 + tn * 16 + l16;
                size_t idx = (size_t)row * N + col;
                float v = acc[tm][tn][r] + bias[col];
                if (doGelu) v = 0.5f * v * (1.0f + erff(v * 0.70710678118654752f));
                if (residual) v += residual[idx];
                if (outF) outF[idx] = v;
                if (outB) outB[idx] = (__bf16)v;
            }
}

// ---------------------------------------------------------------------------
// Flash attention with additive bias, bf16 WMMA, f32 online softmax.
//   Grid: (SEQ/128, B*NH).  Block: 256 = 8 waves; wave w owns 16 query rows.
//   K tile arrives via async global->LDS; V is loaded + transposed in LDS.
// ---------------------------------------------------------------------------
#define Q_BLK  128
#define K_TILE 64
#define K_LD   72   // 144B row stride: 16B-aligned, 36-bank step (conflict-free)

__launch_bounds__(256)
__global__ void attn_kernel(const __bf16* __restrict__ Q,
                            const __bf16* __restrict__ Km,
                            const __bf16* __restrict__ V,
                            const float*  __restrict__ bias,   // [NH,SEQ,SEQ]
                            __bf16* __restrict__ O)            // [B*SEQ, H_DIM] bf16
{
    __shared__ __align__(16) __bf16 Klds[K_TILE * K_LD];   // [key][d]
    __shared__ __align__(16) __bf16 Vt  [HDIM   * K_LD];   // [d][key]
    __shared__ __align__(16) __bf16 Plds[8 * 16 * K_LD];   // per-wave 16 x 64

    const int bh   = blockIdx.y;
    const int b    = bh / NHEAD;
    const int h    = bh % NHEAD;
    const int q0   = blockIdx.x * Q_BLK;
    const int tid  = threadIdx.x;
    const int lane = tid & 31;
    const int wave = tid >> 5;
    const int halfw = lane >> 4;
    const int l16   = lane & 15;

    // Q fragments (registers for all 16 key tiles)
    Frag qf[2];
    {
        size_t qrow = (size_t)b * SEQ + q0 + wave * 16 + l16;
        const __bf16* qp = Q + qrow * H_DIM + h * HDIM;
        for (int kf = 0; kf < 2; ++kf)
            for (int r = 0; r < 8; ++r) {
                int k = kf * 32 + 16 * (r >> 2) + 8 * halfw + 2 * (r & 3);
                qf[kf].u[r] = *(const uint32_t*)(qp + k);
            }
    }

    float mrow[8], lrow[8];
    v8f o[4];
    for (int r = 0; r < 8; ++r) { mrow[r] = -3.0e38f; lrow[r] = 0.0f; }
    for (int tn = 0; tn < 4; ++tn)
        for (int e = 0; e < 8; ++e) o[tn][e] = 0.0f;

    __bf16* myP = &Plds[wave * 16 * K_LD];

    for (int kt = 0; kt < SEQ / K_TILE; ++kt) {
        const int keyBase = kt * K_TILE;
        // ---- K tile via async copy; V tile loaded + transposed ----
        for (int it = 0; it < 2; ++it) {
            int c   = tid + it * 256;
            int row = c >> 3;             // key (8 chunks per 128B row)
            int col = (c & 7) * 8;        // d
            size_t src = ((size_t)b * SEQ + keyBase + row) * H_DIM + h * HDIM + col;
            uint32_t lo = lds_off(&Klds[row * K_LD + col]);
            const __bf16* gp = Km + src;
            asm volatile("global_load_async_to_lds_b128 %0, %1, off"
                         :: "v"(lo), "v"(gp) : "memory");
            uint4 vv = *reinterpret_cast<const uint4*>(V + src);
            union { uint4 v; __bf16 h[8]; } t; t.v = vv;
            for (int i = 0; i < 8; ++i)
                Vt[(col + i) * K_LD + row] = t.h[i];
        }
        wait_async_all();
        __syncthreads();

        // ---- S = Q @ K^T ----
        v8f s[4];
        for (int tn = 0; tn < 4; ++tn) {
            for (int e = 0; e < 8; ++e) s[tn][e] = 0.0f;
            int key = tn * 16 + l16;
            Frag b0, b1;
            for (int r = 0; r < 8; ++r) {
                int k = 16 * (r >> 2) + 8 * halfw + 2 * (r & 3);
                b0.u[r] = *(const uint32_t*)&Klds[key * K_LD + k];        // d 0..31
                b1.u[r] = *(const uint32_t*)&Klds[key * K_LD + 32 + k];   // d 32..63
            }
            s[tn] = wmma_bf16(qf[0], b0, s[tn]);
            s[tn] = wmma_bf16(qf[1], b1, s[tn]);
        }

        // ---- scale + bias + online softmax ----
        for (int r = 0; r < 8; ++r) {
            int qr = q0 + wave * 16 + 8 * halfw + r;
            const float* bp = bias + (size_t)h * SEQ * SEQ + (size_t)qr * SEQ + keyBase;
            float tmax = -3.0e38f;
            for (int tn = 0; tn < 4; ++tn) {
                s[tn][r] = s[tn][r] * 0.125f + bp[tn * 16 + l16];
                tmax = fmaxf(tmax, s[tn][r]);
            }
            for (int d = 1; d < 16; d <<= 1)
                tmax = fmaxf(tmax, __shfl_xor(tmax, d, 32));
            float mnew  = fmaxf(mrow[r], tmax);
            float alpha = expf(mrow[r] - mnew);
            float psum  = 0.0f;
            for (int tn = 0; tn < 4; ++tn) {
                float p = expf(s[tn][r] - mnew);
                s[tn][r] = p;
                psum += p;
            }
            for (int d = 1; d < 16; d <<= 1)
                psum += __shfl_xor(psum, d, 32);
            lrow[r] = lrow[r] * alpha + psum;
            mrow[r] = mnew;
            for (int tn = 0; tn < 4; ++tn) o[tn][r] *= alpha;
            for (int tn = 0; tn < 4; ++tn)
                myP[(8 * halfw + r) * K_LD + tn * 16 + l16] = (__bf16)s[tn][r];
        }

        // ---- O += P @ V ----
        Frag pf[2];
        for (int kf = 0; kf < 2; ++kf)
            for (int r = 0; r < 8; ++r) {
                int k = kf * 32 + 16 * (r >> 2) + 8 * halfw + 2 * (r & 3);
                pf[kf].u[r] = *(const uint32_t*)&myP[l16 * K_LD + k];
            }
        for (int tn = 0; tn < 4; ++tn) {
            int dcol = tn * 16 + l16;
            Frag b0, b1;
            for (int r = 0; r < 8; ++r) {
                int k = 16 * (r >> 2) + 8 * halfw + 2 * (r & 3);
                b0.u[r] = *(const uint32_t*)&Vt[dcol * K_LD + k];         // keys 0..31
                b1.u[r] = *(const uint32_t*)&Vt[dcol * K_LD + 32 + k];    // keys 32..63
            }
            o[tn] = wmma_bf16(pf[0], b0, o[tn]);
            o[tn] = wmma_bf16(pf[1], b1, o[tn]);
        }
        __syncthreads();   // protect Klds/Vt for next tile
    }

    for (int tn = 0; tn < 4; ++tn)
        for (int r = 0; r < 8; ++r) {
            int row = q0 + wave * 16 + 8 * halfw + r;
            int col = h * HDIM + tn * 16 + l16;
            float ov = o[tn][r] / lrow[r];
            O[((size_t)b * SEQ + row) * H_DIM + col] = (__bf16)ov;
        }
}

// ---------------------------------------------------------------------------
// LayerNorm over H_DIM=768; one block per row. Biased variance + eps 1e-5.
// ---------------------------------------------------------------------------
__launch_bounds__(256)
__global__ void layernorm_kernel(const float* __restrict__ in,
                                 const float* __restrict__ g,
                                 const float* __restrict__ be,
                                 float* __restrict__ outF,
                                 __bf16* __restrict__ outB)   // nullable
{
    __shared__ float rs[256], rs2[256];
    const int tid = threadIdx.x;
    const float* x = in + (size_t)blockIdx.x * H_DIM;
    float s = 0.0f, s2 = 0.0f;
    for (int i = tid; i < H_DIM; i += 256) {
        float v = x[i];
        s += v; s2 += v * v;
    }
    rs[tid] = s; rs2[tid] = s2;
    __syncthreads();
    for (int off = 128; off > 0; off >>= 1) {
        if (tid < off) { rs[tid] += rs[tid + off]; rs2[tid] += rs2[tid + off]; }
        __syncthreads();
    }
    float mean = rs[0] * (1.0f / H_DIM);
    float var  = rs2[0] * (1.0f / H_DIM) - mean * mean;
    float rstd = rsqrtf(var + 1e-5f);
    float* of  = outF + (size_t)blockIdx.x * H_DIM;
    for (int i = tid; i < H_DIM; i += 256) {
        float v = (x[i] - mean) * rstd * g[i] + be[i];
        of[i] = v;
        if (outB) outB[(size_t)blockIdx.x * H_DIM + i] = (__bf16)v;
    }
}

// ---------------------------------------------------------------------------
// Host orchestration
// ---------------------------------------------------------------------------
extern "C" void kernel_launch(void* const* d_in, const int* in_sizes, int n_in,
                              void* d_out, int out_size, void* d_ws, size_t ws_size,
                              hipStream_t stream)
{
    (void)in_sizes; (void)n_in; (void)out_size; (void)ws_size;

    const float* x   = (const float*)d_in[0];
    const float* ab  = (const float*)d_in[1];
    const float* Wq  = (const float*)d_in[2];
    const float* bq  = (const float*)d_in[3];
    const float* Wk  = (const float*)d_in[4];
    const float* bk  = (const float*)d_in[5];
    const float* Wv  = (const float*)d_in[6];
    const float* bv  = (const float*)d_in[7];
    const float* Wo  = (const float*)d_in[8];
    const float* bo  = (const float*)d_in[9];
    const float* g1  = (const float*)d_in[10];
    const float* be1 = (const float*)d_in[11];
    const float* W1  = (const float*)d_in[12];
    const float* bf1 = (const float*)d_in[13];
    const float* W2  = (const float*)d_in[14];
    const float* bf2 = (const float*)d_in[15];
    const float* g2  = (const float*)d_in[16];
    const float* be2 = (const float*)d_in[17];

    char* ws = (char*)d_ws;
    size_t off = 0;
    auto alloc = [&](size_t bytes) -> void* {
        void* p = ws + off;
        off += (bytes + 255) & ~(size_t)255;
        return p;
    };

    const size_t NE_X  = (size_t)MROWS * H_DIM;
    const size_t NE_HH = (size_t)H_DIM * H_DIM;
    const size_t NE_HF = (size_t)H_DIM * FF_DIM;

    __bf16* xb  = (__bf16*)alloc(NE_X  * sizeof(__bf16));
    __bf16* wqb = (__bf16*)alloc(NE_HH * sizeof(__bf16));   // transposed [N][K]
    __bf16* wkb = (__bf16*)alloc(NE_HH * sizeof(__bf16));
    __bf16* wvb = (__bf16*)alloc(NE_HH * sizeof(__bf16));
    __bf16* wob = (__bf16*)alloc(NE_HH * sizeof(__bf16));
    __bf16* w1b = (__bf16*)alloc(NE_HF * sizeof(__bf16));
    __bf16* w2b = (__bf16*)alloc(NE_HF * sizeof(__bf16));
    __bf16* qb  = (__bf16*)alloc(NE_X  * sizeof(__bf16));
    __bf16* kb  = (__bf16*)alloc(NE_X  * sizeof(__bf16));
    __bf16* vb  = (__bf16*)alloc(NE_X  * sizeof(__bf16));
    __bf16* aob = (__bf16*)alloc(NE_X  * sizeof(__bf16));
    float*  y1  = (float*) alloc(NE_X  * sizeof(float));
    float*  x1f = (float*) alloc(NE_X  * sizeof(float));
    __bf16* x1b = (__bf16*)alloc(NE_X  * sizeof(__bf16));
    __bf16* hb  = (__bf16*)alloc((size_t)MROWS * FF_DIM * sizeof(__bf16));
    float*  y2  = (float*) alloc(NE_X  * sizeof(float));

    // activations: straight convert
    {
        int blocks = (int)((NE_X + 255) / 256);
        cvt_f32_bf16_kernel<<<blocks, 256, 0, stream>>>(x, xb, (int)NE_X);
    }
    // weights: convert + transpose ([K][N] -> [N][K])
    auto cvtT = [&](const float* src, __bf16* dst, int K, int N) {
        dim3 g(N / 32, K / 32);
        cvt_transpose_kernel<<<g, 256, 0, stream>>>(src, dst, K, N);
    };
    cvtT(Wq, wqb, H_DIM, H_DIM);
    cvtT(Wk, wkb, H_DIM, H_DIM);
    cvtT(Wv, wvb, H_DIM, H_DIM);
    cvtT(Wo, wob, H_DIM, H_DIM);
    cvtT(W1, w1b, H_DIM, FF_DIM);
    cvtT(W2, w2b, FF_DIM, H_DIM);

    // ---- QKV projections (bf16 out for attention) ----
    dim3 gHH(H_DIM / TB_N, MROWS / TB_M);
    gemm_bf16_kernel<<<gHH, 256, 0, stream>>>(xb, wqb, bq, nullptr, nullptr, qb,
                                              MROWS, H_DIM, H_DIM, 0);
    gemm_bf16_kernel<<<gHH, 256, 0, stream>>>(xb, wkb, bk, nullptr, nullptr, kb,
                                              MROWS, H_DIM, H_DIM, 0);
    gemm_bf16_kernel<<<gHH, 256, 0, stream>>>(xb, wvb, bv, nullptr, nullptr, vb,
                                              MROWS, H_DIM, H_DIM, 0);

    // ---- flash attention with bias ----
    dim3 gAttn(SEQ / Q_BLK, BATCH * NHEAD);
    attn_kernel<<<gAttn, 256, 0, stream>>>(qb, kb, vb, ab, aob);

    // ---- output projection + residual(x) -> y1 ; LN1 -> x1 (f32 + bf16) ----
    gemm_bf16_kernel<<<gHH, 256, 0, stream>>>(aob, wob, bo, x, y1, nullptr,
                                              MROWS, H_DIM, H_DIM, 0);
    layernorm_kernel<<<MROWS, 256, 0, stream>>>(y1, g1, be1, x1f, x1b);

    // ---- FFN: GELU(x1 @ W1 + bf1) @ W2 + bf2 + residual(x1) -> y2 ; LN2 ----
    dim3 gHF(FF_DIM / TB_N, MROWS / TB_M);
    gemm_bf16_kernel<<<gHF, 256, 0, stream>>>(x1b, w1b, bf1, nullptr, nullptr, hb,
                                              MROWS, H_DIM, FF_DIM, 1);
    gemm_bf16_kernel<<<gHH, 256, 0, stream>>>(hb, w2b, bf2, x1f, y2, nullptr,
                                              MROWS, FF_DIM, H_DIM, 0);
    layernorm_kernel<<<MROWS, 256, 0, stream>>>(y2, g2, be2, (float*)d_out, nullptr);
}
